// SummaRunnerSentenceExtractor_9053791060556
// MI455X (gfx1250) — matrix operand order repaired and checked by reading, not compiled, not run on hardware
//
#include <hip/hip_runtime.h>
#include <hip/hip_bf16.h>
#include <math.h>

typedef __attribute__((ext_vector_type(16))) _Float16 v16h;
typedef __attribute__((ext_vector_type(8)))  _Float16 v8h;
typedef __attribute__((ext_vector_type(8)))  float    v8f;

// Problem constants
#define BB   256
#define SS   50
#define II   1024
#define HH   512
#define SENTD 512
#define DOCD  1024
#define SEGMENTS 4
#define MAXPOS   25
#define POSSZ    50
#define SEGSZ    50

// A-matrix gather modes for the generic GEMM
#define A_PLAIN  0   // A[m*K + k]
#define A_XF     1   // x time-major forward: m = s*B+b -> x[(b*S+s)*K + k]
#define A_XB     2   // x time-major reversed: m = s*B+b -> x[(b*S+(S-1-s))*K + k]
#define A_CONCAT 3   // m = b*S+s ; k<H -> hsF[(s*B+b)*H+k] else hsB[(s*B+b)*H+k-H]
#define A_TANH   4   // tanhf(A[m*K + k])

#define ACT_NONE 0
#define ACT_RELU 1
#define ACT_TANH 2

// GEMM tiling: block tile 128(M) x 128(N), K step 32, 8 waves of 32 lanes.
// Waves arranged 4(M) x 2(N); each wave owns a 32x64 sub-tile = 2x4 WMMA tiles.
#define MT 128
#define NT 128
#define KT 32
#define LST 40   // padded LDS row stride in f16 elements (80B, 16B-aligned rows)

__device__ __forceinline__ float loadA(int mode, const float* __restrict__ A,
                                       const float* __restrict__ A2,
                                       int m, int k, int K) {
  switch (mode) {
    case A_PLAIN: return A[(size_t)m * K + k];
    case A_XF: { int s = m / BB, b = m % BB;
                 return A[((size_t)b * SS + s) * K + k]; }
    case A_XB: { int s = m / BB, b = m % BB;
                 return A[((size_t)b * SS + (SS - 1 - s)) * K + k]; }
    case A_CONCAT: { int b = m / SS, s = m % SS;
                 return (k < HH) ? A [((size_t)s * BB + b) * HH + k]
                                 : A2[((size_t)s * BB + b) * HH + (k - HH)]; }
    case A_TANH: return tanhf(A[(size_t)m * K + k]);
  }
  return 0.f;
}

__device__ __forceinline__ v16h frag16(const _Float16* row, int off0, int off1) {
  v8h lo = *(const v8h*)(row + off0);
  v8h hi = *(const v8h*)(row + off1);
  return __builtin_shufflevector(lo, hi, 0, 1, 2, 3, 4, 5, 6, 7,
                                 8, 9, 10, 11, 12, 13, 14, 15);
}

// Generic LDS-tiled GEMM: out[m,n] = act( sum_k A(m,k)*W[n*K+k] + bias[n] )
// Requires M%128==0, N%128==0, K%32==0 (true for every GEMM in this model).
__global__ void __launch_bounds__(256)
wmma_gemm(const float* __restrict__ A, const float* __restrict__ A2,
          const float* __restrict__ W, const float* __restrict__ bias,
          float* __restrict__ out, int M, int N, int K, int mode, int act) {
  __shared__ _Float16 aT[MT * LST];
  __shared__ _Float16 bT[NT * LST];

  const int lane = threadIdx.x & 31;
  const int wave = threadIdx.x >> 5;
  const int wm = wave >> 1;        // 0..3 : M sub-block of 32 rows
  const int wn = wave & 1;         // 0..1 : N sub-block of 64 cols
  const int hsel = lane >> 4;
  const int l16  = lane & 15;

  const int nBlocks = N / NT;
  const int bn = blockIdx.x % nBlocks;
  const int bm = blockIdx.x / nBlocks;
  const int m0 = bm * MT, n0 = bn * NT;

  v8f c[2][4] = {};

  for (int kc = 0; kc < K; kc += KT) {
    // Cooperative fill: 256 threads x 16 elements each covers 128x32 per tile.
    // Consecutive threads touch consecutive k -> coalesced global loads.
#pragma unroll
    for (int t = 0; t < 16; ++t) {
      int i  = threadIdx.x + t * 256;   // 0..4095
      int rl = i >> 5;                  // 0..127
      int kl = i & 31;                  // 0..31
      aT[rl * LST + kl] = (_Float16)loadA(mode, A, A2, m0 + rl, kc + kl, K);
      bT[rl * LST + kl] = (_Float16)W[(size_t)(n0 + rl) * K + kc + kl];
    }
    __syncthreads();

    // Fragments straight from LDS as 16B vector loads (ISA f16 layouts):
    //   A frag lane: row M=l16, k = hsel*8..+7 and 16+hsel*8..+7  (two b128)
    //   B frag lane: col N=l16, k = hsel*8..+7 and 16+hsel*8..+7  (two b128)
    v16h af[2], bf[4];
#pragma unroll
    for (int i = 0; i < 2; ++i) {
      const _Float16* row = aT + (size_t)(wm * 32 + i * 16 + l16) * LST;
      af[i] = frag16(row, hsel * 8, 16 + hsel * 8);
    }
#pragma unroll
    for (int j = 0; j < 4; ++j) {
      const _Float16* row = bT + (size_t)(wn * 64 + j * 16 + l16) * LST;
      bf[j] = frag16(row, hsel * 8, 16 + hsel * 8);
    }
#pragma unroll
    for (int i = 0; i < 2; ++i)
#pragma unroll
      for (int j = 0; j < 4; ++j)
        c[i][j] = __builtin_amdgcn_wmma_f32_16x16x32_f16(
            false, af[i], false, bf[j], (short)0, c[i][j], false, false);
    __syncthreads();
  }

  // Writeback: C/D layout -> lane l16 = n, reg r -> m = r + hsel*8.
#pragma unroll
  for (int i = 0; i < 2; ++i) {
#pragma unroll
    for (int j = 0; j < 4; ++j) {
      const int n = n0 + wn * 64 + j * 16 + l16;
      const float bv = bias ? bias[n] : 0.f;
#pragma unroll
      for (int r = 0; r < 8; ++r) {
        const int m = m0 + wm * 32 + i * 16 + r + hsel * 8;
        float v = c[i][j][r] + bv;
        if (act == ACT_RELU)      v = fmaxf(v, 0.f);
        else if (act == ACT_TANH) v = tanhf(v);
        out[(size_t)m * N + n] = v;
      }
    }
  }
}

__global__ void zero_f32(float* p, int n) {
  int i = blockIdx.x * blockDim.x + threadIdx.x;
  if (i < n) p[i] = 0.f;
}

// GRU gate elementwise: xg already has b_ih folded in; hg has b_hh folded in.
__global__ void gru_gate(const float* __restrict__ xg, const float* __restrict__ hg,
                         float* __restrict__ h, float* __restrict__ hs,
                         int s, int store_s) {
  int i = blockIdx.x * blockDim.x + threadIdx.x;
  if (i >= BB * HH) return;
  int b = i / HH, j = i % HH;
  size_t xb = ((size_t)s * BB + b) * (3 * HH);
  size_t hb = (size_t)b * (3 * HH);
  float xr = xg[xb + j], xz = xg[xb + HH + j], xn = xg[xb + 2 * HH + j];
  float hr = hg[hb + j], hz = hg[hb + HH + j], hn = hg[hb + 2 * HH + j];
  float r = 1.f / (1.f + expf(-(xr + hr)));
  float z = 1.f / (1.f + expf(-(xz + hz)));
  float n = tanhf(xn + r * hn);
  float hv = (1.f - z) * n + z * h[(size_t)b * HH + j];
  h[(size_t)b * HH + j] = hv;
  hs[((size_t)store_s * BB + b) * HH + j] = hv;
}

// content[m] = sent[m,:] . w_content + b_content   (one wave per row)
__global__ void content_kernel(const float* __restrict__ sent, const float* __restrict__ w,
                               const float* __restrict__ b0, float* __restrict__ out,
                               int rows, int D) {
  int wid  = (blockIdx.x * blockDim.x + threadIdx.x) >> 5;
  int lane = threadIdx.x & 31;
  if (wid >= rows) return;
  float acc = 0.f;
  for (int d = lane; d < D; d += 32) acc += sent[(size_t)wid * D + d] * w[d];
  for (int off = 16; off; off >>= 1) acc += __shfl_down(acc, off, 32);
  if (lane == 0) out[wid] = acc + b0[0];
}

__global__ void avg_kernel(const float* __restrict__ sent, const int* __restrict__ nsent,
                           float* __restrict__ avg) {
  int i = blockIdx.x * blockDim.x + threadIdx.x;
  if (i >= BB * SENTD) return;
  int b = i / SENTD, d = i % SENTD;
  float s = 0.f;
  for (int t = 0; t < SS; ++t) s += sent[((size_t)b * SS + t) * SENTD + d];
  avg[i] = s / (float)nsent[b];
}

// static[s,b] = content + salience + pos_logit(s) + seg_logit(b,s) + bias
__global__ void static_kernel(const float* __restrict__ sent, const float* __restrict__ doc,
                              const float* __restrict__ content,
                              const float* __restrict__ pos_emb, const float* __restrict__ pos_lin,
                              const float* __restrict__ seg_emb, const float* __restrict__ seg_lin,
                              const float* __restrict__ bias, const int* __restrict__ nsent,
                              float* __restrict__ stat) {
  int wid  = (blockIdx.x * blockDim.x + threadIdx.x) >> 5;
  int lane = threadIdx.x & 31;
  if (wid >= BB * SS) return;
  int b = wid / SS, s = wid % SS;
  float acc = 0.f;
  for (int d = lane; d < SENTD; d += 32)
    acc += sent[((size_t)b * SS + s) * SENTD + d] * doc[(size_t)b * SENTD + d];
  for (int off = 16; off; off >>= 1) acc += __shfl_down(acc, off, 32);
  if (lane == 0) {
    int ai = s + 1; if (ai > MAXPOS) ai = MAXPOS;
    float pl = 0.f;
    for (int d = 0; d < POSSZ; ++d) pl += pos_emb[ai * POSSZ + d] * pos_lin[d];
    float chunk = rintf((float)nsent[b] / (float)SEGMENTS);   // round-half-even like jnp.round
    float relf  = ceilf((float)(s + 1) / chunk);
    relf = fminf(fmaxf(relf, 0.f), (float)SEGMENTS);
    int rel = (int)relf;
    float sl = 0.f;
    for (int d = 0; d < SEGSZ; ++d) sl += seg_emb[rel * SEGSZ + d] * seg_lin[d];
    stat[(size_t)s * BB + b] = content[(size_t)b * SS + s] + acc + pl + sl + bias[0];
  }
}

// One scan step: sim = s_t . q ; logit = static - sim ; summary += s_t*sigmoid(logit)
__global__ void scan_step(const float* __restrict__ sent, const float* __restrict__ q,
                          const float* __restrict__ stat, float* __restrict__ summary,
                          float* __restrict__ out, int t) {
  int wid  = (blockIdx.x * blockDim.x + threadIdx.x) >> 5;
  int lane = threadIdx.x & 31;
  if (wid >= BB) return;
  int b = wid;
  const float* st = &sent[((size_t)b * SS + t) * SENTD];
  float acc = 0.f;
  for (int d = lane; d < SENTD; d += 32) acc += st[d] * q[(size_t)b * SENTD + d];
  for (int off = 16; off; off >>= 1) acc += __shfl_xor(acc, off, 32);  // all lanes get sum
  float logit = stat[(size_t)t * BB + b] - acc;
  float sg = 1.f / (1.f + expf(-logit));
  for (int d = lane; d < SENTD; d += 32) summary[(size_t)b * SENTD + d] += st[d] * sg;
  if (lane == 0) out[(size_t)t * BB + b] = logit;
}

extern "C" void kernel_launch(void* const* d_in, const int* in_sizes, int n_in,
                              void* d_out, int out_size, void* d_ws, size_t ws_size,
                              hipStream_t stream) {
  const float* x        = (const float*)d_in[0];
  const int*   nsent    = (const int*)  d_in[1];
  const float* W_ih_f   = (const float*)d_in[2];
  const float* W_hh_f   = (const float*)d_in[3];
  const float* b_ih_f   = (const float*)d_in[4];
  const float* b_hh_f   = (const float*)d_in[5];
  const float* W_ih_b   = (const float*)d_in[6];
  const float* W_hh_b   = (const float*)d_in[7];
  const float* b_ih_b   = (const float*)d_in[8];
  const float* b_hh_b   = (const float*)d_in[9];
  const float* W_sent   = (const float*)d_in[10];
  const float* b_sent   = (const float*)d_in[11];
  const float* w_cont   = (const float*)d_in[12];
  const float* b_cont   = (const float*)d_in[13];
  const float* W_d1     = (const float*)d_in[14];
  const float* b_d1     = (const float*)d_in[15];
  const float* W_d2     = (const float*)d_in[16];
  const float* b_d2     = (const float*)d_in[17];
  const float* W_sim    = (const float*)d_in[18];
  const float* bias     = (const float*)d_in[19];
  const float* pos_emb  = (const float*)d_in[20];
  const float* pos_lin  = (const float*)d_in[21];
  const float* seg_emb  = (const float*)d_in[22];
  const float* seg_lin  = (const float*)d_in[23];
  float* out = (float*)d_out;   // [S,B,1] -> out[s*B + b]

  // Workspace carve-up (floats)
  float* ws = (float*)d_ws;
  size_t off = 0;
  auto carve = [&](size_t n) { float* p = ws + off; off += n; return p; };
  const size_t M_SB = (size_t)SS * BB;                 // 12800
  float* xg_f    = carve(M_SB * 3 * HH);
  float* xg_b    = carve(M_SB * 3 * HH);
  float* hs_f    = carve(M_SB * HH);
  float* hs_b    = carve(M_SB * HH);
  float* h_f     = carve((size_t)BB * HH);
  float* h_b     = carve((size_t)BB * HH);
  float* hg_f    = carve((size_t)BB * 3 * HH);
  float* hg_b    = carve((size_t)BB * 3 * HH);
  float* sent    = carve(M_SB * SENTD);
  float* content = carve(M_SB);
  float* avg     = carve((size_t)BB * SENTD);
  float* t1      = carve((size_t)BB * DOCD);
  float* doc     = carve((size_t)BB * SENTD);
  float* stat    = carve(M_SB);
  float* summary = carve((size_t)BB * SENTD);
  float* qbuf    = carve((size_t)BB * SENTD);
  (void)ws_size;

  auto gemm = [&](const float* A, const float* A2, const float* W, const float* bvec,
                  float* o, int M, int N, int K, int mode, int act) {
    int blocks = (M / MT) * (N / NT);
    wmma_gemm<<<blocks, 256, 0, stream>>>(A, A2, W, bvec, o, M, N, K, mode, act);
  };

  // Zero-init recurrent state each call (deterministic across graph replays)
  zero_f32<<<((BB * HH) + 255) / 256, 256, 0, stream>>>(h_f, BB * HH);
  zero_f32<<<((BB * HH) + 255) / 256, 256, 0, stream>>>(h_b, BB * HH);
  zero_f32<<<((BB * SENTD) + 255) / 256, 256, 0, stream>>>(summary, BB * SENTD);

  // 1) Input gates for both directions (big compute-bound GEMMs)
  gemm(x, nullptr, W_ih_f, b_ih_f, xg_f, (int)M_SB, 3 * HH, II, A_XF, ACT_NONE);
  gemm(x, nullptr, W_ih_b, b_ih_b, xg_b, (int)M_SB, 3 * HH, II, A_XB, ACT_NONE);

  // 2) GRU recurrence, 50 sequential steps per direction
  for (int t = 0; t < SS; ++t) {
    gemm(h_f, nullptr, W_hh_f, b_hh_f, hg_f, BB, 3 * HH, HH, A_PLAIN, ACT_NONE);
    gru_gate<<<((BB * HH) + 255) / 256, 256, 0, stream>>>(xg_f, hg_f, h_f, hs_f, t, t);
    gemm(h_b, nullptr, W_hh_b, b_hh_b, hg_b, BB, 3 * HH, HH, A_PLAIN, ACT_NONE);
    gru_gate<<<((BB * HH) + 255) / 256, 256, 0, stream>>>(xg_b, hg_b, h_b, hs_b, t, SS - 1 - t);
  }

  // 3) Sentence representations: ReLU(concat(hs_f,hs_b) @ W_sent^T + b)
  gemm(hs_f, hs_b, W_sent, b_sent, sent, (int)M_SB, SENTD, 2 * HH, A_CONCAT, ACT_RELU);

  // 4) Content logits
  content_kernel<<<(((int)M_SB * 32) + 255) / 256, 256, 0, stream>>>(
      sent, w_cont, b_cont, content, (int)M_SB, SENTD);

  // 5) Document representation
  avg_kernel<<<((BB * SENTD) + 255) / 256, 256, 0, stream>>>(sent, nsent, avg);
  gemm(avg, nullptr, W_d1, b_d1, t1,  BB, DOCD,  SENTD, A_PLAIN, ACT_TANH);
  gemm(t1,  nullptr, W_d2, b_d2, doc, BB, SENTD, DOCD,  A_PLAIN, ACT_NONE);

  // 6) Static logits (salience + content + position + segment + bias)
  static_kernel<<<(((int)M_SB * 32) + 255) / 256, 256, 0, stream>>>(
      sent, doc, content, pos_emb, pos_lin, seg_emb, seg_lin, bias, nsent, stat);

  // 7) Sequential novelty scan: q = tanh(summary) @ W_sim^T, then update
  for (int t = 0; t < SS; ++t) {
    gemm(summary, nullptr, W_sim, nullptr, qbuf, BB, SENTD, SENTD, A_TANH, ACT_NONE);
    scan_step<<<((BB * 32) + 255) / 256, 256, 0, stream>>>(sent, qbuf, stat, summary, out, t);
  }
}
